// RSCN_10548439679058
// MI455X (gfx1250) — matrix operand order repaired and checked by reading, not compiled, not run on hardware
//
#include <hip/hip_runtime.h>
#include <hip/hip_bf16.h>
#include <math.h>

// ---------------- problem constants (match reference shapes) ----------------
#define B_   256
#define T_   512
#define D_   256
#define H_   512
#define S_   8
#define NG_  2048          // 4*H
#define NGROUPS 8          // batch groups
#define MB   32            // batch rows per group (B/NGROUPS)
#define WPG  16            // workgroups per group (one per 32 hidden units)
#define THREADS 256

typedef __bf16 bf16;
typedef bf16  v16bf  __attribute__((ext_vector_type(16)));
typedef bf16  bf16x8 __attribute__((ext_vector_type(8)));
typedef float v8f    __attribute__((ext_vector_type(8)));

__device__ inline v16bf cat16(bf16x8 lo, bf16x8 hi) {
  return __builtin_shufflevector(lo, hi, 0,1,2,3,4,5,6,7,8,9,10,11,12,13,14,15);
}
__device__ inline v8f wmma_bf16(v16bf a, v16bf b, v8f c) {
  return __builtin_amdgcn_wmma_f32_16x16x32_bf16(false, a, false, b, (short)0, c, false, false);
}

// ---- CDNA5 async global->LDS DMA (ASYNCcnt-tracked), per CDNA5 ISA ----
__device__ inline unsigned lds_addr_of(const void* p) {
  // generic shared-aperture pointer: low 32 bits are the DS byte address
  return (unsigned)(unsigned long long)p;
}
__device__ inline void async_ld_b128(unsigned lds_addr, const void* gaddr) {
  asm volatile("global_load_async_to_lds_b128 %0, %1, off"
               :: "v"(lds_addr), "v"(gaddr) : "memory");
}
__device__ inline void wait_async0() {
  asm volatile("s_wait_asynccnt 0x0" ::: "memory");
}

// ---------------- setup kernels ----------------

__global__ void k_zero(unsigned* ctr) {
  for (int i = threadIdx.x; i < NGROUPS * 64; i += THREADS) ctr[i] = 0u;
}

// W_eff^T[n][k] = W_hh[n][k] + sum_d W_lin[d][k] * W_ih[n][d]   (stored bf16, n-major)
// b_eff[n]     = b_ih[n] + b_hh[n] + sum_d b_lin[d] * W_ih[n][d]
__global__ void k_weff(const float* __restrict__ W_ih, const float* __restrict__ W_hh,
                       const float* __restrict__ W_lin, const float* __restrict__ b_ih,
                       const float* __restrict__ b_hh, const float* __restrict__ b_lin,
                       bf16* __restrict__ WeffT, float* __restrict__ beff) {
  __shared__ float sx[D_];
  const int n = blockIdx.x;
  sx[threadIdx.x] = W_ih[n * D_ + threadIdx.x];
  __syncthreads();
  for (int k = threadIdx.x; k < H_; k += THREADS) {
    float acc = W_hh[n * H_ + k];
    for (int d = 0; d < D_; ++d) acc += W_lin[d * H_ + k] * sx[d];
    WeffT[(size_t)n * H_ + k] = (bf16)acc;
  }
  if (threadIdx.x == 0) {
    float acc = b_ih[n] + b_hh[n];
    for (int d = 0; d < D_; ++d) acc += b_lin[d] * sx[d];
    beff[n] = acc;
  }
}

__global__ void k_wlin(const float* __restrict__ W_lin, bf16* __restrict__ WlinT) {
  const int idx = blockIdx.x * THREADS + threadIdx.x;   // 256*512 elems, identity layout
  if (idx < D_ * H_) WlinT[idx] = (bf16)W_lin[idx];
}

// gates0[b][n] = x0[b]@W_ih^T + h0[b]@W_hh^T + b_ih + b_hh  (step-0 special case)
__global__ void k_gates0(const float* __restrict__ inp, const float* __restrict__ h0,
                         const float* __restrict__ W_ih, const float* __restrict__ W_hh,
                         const float* __restrict__ b_ih, const float* __restrict__ b_hh,
                         float* __restrict__ gates0) {
  __shared__ float sx[D_];
  __shared__ float sh[H_];
  const int b = blockIdx.x;
  sx[threadIdx.x] = inp[(size_t)b * T_ * D_ + threadIdx.x];     // _input[b][0][:]
  sh[threadIdx.x]       = h0[b * H_ + threadIdx.x];
  sh[threadIdx.x + 256] = h0[b * H_ + threadIdx.x + 256];
  __syncthreads();
  for (int n = threadIdx.x; n < NG_; n += THREADS) {
    float acc = b_ih[n] + b_hh[n];
    for (int d = 0; d < D_; ++d) acc += W_ih[(size_t)n * D_ + d] * sx[d];
    for (int k = 0; k < H_; ++k)  acc += W_hh[(size_t)n * H_ + k] * sh[k];
    gates0[(size_t)b * NG_ + n] = acc;
  }
}

// ---------------- main persistent kernel ----------------
// grid = NGROUPS*WPG = 128 blocks of 256 threads; one block per WGP (~300KB LDS).
// Group g handles batch rows [g*32, g*32+32); WG j of the group owns hidden
// units [32j,32j+32) (gate cols {u,512+u,1024+u,1536+u}) and out cols [16j,16j+16).

__device__ inline void group_barrier(unsigned* c, unsigned seq) {
  __builtin_amdgcn_fence(__ATOMIC_RELEASE, "agent");
  __syncthreads();
  if (threadIdx.x == 0) {
    __hip_atomic_fetch_add(c, 1u, __ATOMIC_RELAXED, __HIP_MEMORY_SCOPE_AGENT);
    const unsigned target = (unsigned)WPG * seq;
    while (__hip_atomic_load(c, __ATOMIC_RELAXED, __HIP_MEMORY_SCOPE_AGENT) < target)
      __builtin_amdgcn_s_sleep(2);
  }
  __syncthreads();
  __builtin_amdgcn_fence(__ATOMIC_ACQUIRE, "agent");
}

__global__ __launch_bounds__(THREADS, 1)
void rscn_main(const float* __restrict__ h0, const float* __restrict__ c0,
               const int* __restrict__ w1, const int* __restrict__ w2,
               const float* __restrict__ b_lin,
               const bf16* __restrict__ WeffT, const float* __restrict__ beff,
               const bf16* __restrict__ WlinT, const float* __restrict__ gates0,
               bf16* __restrict__ HcBuf, unsigned* __restrict__ ctr,
               float* __restrict__ out) {
  extern __shared__ char smem_raw[];
  bf16*  sWeff  = (bf16*)smem_raw;        // [128][512] bf16  (n-major weight slice)
  bf16*  sWlin  = sWeff + 128 * 512;      // [16][512]  bf16
  bf16*  sA0    = sWlin + 16 * 512;       // [32][512]  bf16  (comb_h, double-buffered)
  bf16*  sA1    = sA0 + 32 * 512;         // [32][512]  bf16
  float* sGates = (float*)(sA1 + 32 * 512);// [32][128] f32
  float* sBh    = sGates + 32 * 128;      // [8][32][32] f32  (skip ring h)
  float* sBc    = sBh + 8 * 32 * 32;      // [8][32][32] f32  (skip ring c)
  float* sC     = sBc + 8 * 32 * 32;      // [32][32]   f32  (cell state)
  float* sBeff  = sC + 32 * 32;           // [128]      f32

  const int tid   = threadIdx.x;
  const int wave  = tid >> 5;
  const int lane  = tid & 31;
  const int lrow  = lane & 15;
  const bool hi   = lane >= 16;
  const int wg    = blockIdx.x & (WPG - 1);
  const int group = blockIdx.x / WPG;
  const int grow0 = group * MB;
  unsigned* myctr = ctr + group * 64;

  // ---- stage persistent weight slices + step-0 gates via async DMA (16B/lane) ----
  {
    // W_eff slice: 128 rows (global row n(c)) x 1KB each = 8192 chunks of 16B
    const char* src = (const char*)WeffT;
    const unsigned dst = lds_addr_of(sWeff);
    for (int i = tid; i < 128 * 64; i += THREADS) {
      const int c = i >> 6, k16 = i & 63;
      const int n = ((c >> 5) << 9) + wg * 32 + (c & 31);   // global gate col
      async_ld_b128(dst + (unsigned)i * 16, src + (size_t)n * 1024 + k16 * 16);
    }
  }
  {
    // W_lin slice: 16 rows x 1KB = 1024 chunks of 16B
    const char* src = (const char*)WlinT;
    const unsigned dst = lds_addr_of(sWlin);
    for (int i = tid; i < 16 * 64; i += THREADS) {
      const int j = i >> 6, k16 = i & 63;
      async_ld_b128(dst + (unsigned)i * 16, src + (size_t)(wg * 16 + j) * 1024 + k16 * 16);
    }
  }
  {
    // step-0 gates: 32 rows x 128 f32; each 16B chunk stays inside one 32-col gate block
    const char* src = (const char*)gates0;
    const unsigned dst = lds_addr_of(sGates);
    for (int i = tid; i < MB * 32; i += THREADS) {
      const int row = i >> 5, q = (i & 31) * 4;             // local col of chunk
      const int n = ((q >> 5) << 9) + wg * 32 + (q & 31);
      async_ld_b128(dst + (unsigned)i * 16,
                    src + ((size_t)(grow0 + row) * NG_ + n) * 4);
    }
  }
  for (int i = tid; i < 128; i += THREADS) {
    const int n = ((i >> 5) << 9) + wg * 32 + (i & 31);
    sBeff[i] = beff[n];
  }
  // rings / cell state from h0,c0
  for (int i = tid; i < MB * 32; i += THREADS) {
    const int row = i >> 5, u = i & 31;
    const float h0v = h0[(grow0 + row) * H_ + wg * 32 + u];
    const float c0v = c0[(grow0 + row) * H_ + wg * 32 + u];
#pragma unroll
    for (int s = 0; s < S_; ++s) { sBh[s * 1024 + i] = h0v; sBc[s * 1024 + i] = c0v; }
    sC[i] = c0v;
  }
  wait_async0();
  __syncthreads();

  // ---- elementwise (LSTM cell + skip combine), publishes comb_h slice ----
  auto elementwise = [&](int t) {
    const float a = (float)w1[t];
    const float b = (float)w2[t];
    const float norm = 1.0f / (a + b);
    const int pos = t & (S_ - 1);
    bf16* hdst = HcBuf + (size_t)(group * 2 + (t & 1)) * (MB * H_);
    for (int i = tid; i < MB * 32; i += THREADS) {
      const int row = i >> 5, u = i & 31;
      const float gi = sGates[row * 128 + u];
      const float gf = sGates[row * 128 + 32 + u];
      const float gg = sGates[row * 128 + 64 + u];
      const float go = sGates[row * 128 + 96 + u];
      const float si = 1.0f / (1.0f + expf(-gi));
      const float sf = 1.0f / (1.0f + expf(-gf));
      const float so = 1.0f / (1.0f + expf(-go));
      const float cn = sf * sC[i] + si * tanhf(gg);
      const float hn = so * tanhf(cn);
      const int rs = pos * 1024 + i;
      const float sh = sBh[rs], sc = sBc[rs];
      sBh[rs] = hn; sBc[rs] = cn;
      const float ch = (a * hn + b * tanhf(sh)) * norm;
      const float cc = (a * cn + b * tanhf(sc)) * norm;
      sC[i] = cc;
      hdst[row * H_ + wg * 32 + u] = (bf16)ch;
      if (t == T_ - 1) {                       // h_fin / c_fin
        float* hf = out + (size_t)B_ * T_ * D_;
        hf[(grow0 + row) * H_ + wg * 32 + u] = ch;
        hf[B_ * H_ + (grow0 + row) * H_ + wg * 32 + u] = cc;
      }
    }
  };

  // issue async broadcast of full comb_h (32x512 bf16 = 2048 x 16B) into a LDS tile
  auto asyncLoadA = [&](bf16* dstT, int p) {
    const char* src = (const char*)(HcBuf + (size_t)(group * 2 + p) * (MB * H_));
    const unsigned dst = lds_addr_of(dstT);
    for (int i = tid; i < MB * H_ / 8; i += THREADS)
      async_ld_b128(dst + (unsigned)i * 16, src + (size_t)i * 16);
  };

  // gates GEMM: [32 x 512] (bf16) @ W_eff slice [512 x 128] -> sGates, +bias
  auto gatesGemm = [&](const bf16* sA) {
    const int c0col = wave * 16;
    const bf16* bbase  = sWeff + (c0col + lrow) * 512 + (hi ? 16 : 0);
    const bf16* abase0 = sA + lrow * 512 + (hi ? 16 : 0);
    const bf16* abase1 = abase0 + 16 * 512;
    v8f acc0 = {}, acc1 = {};
#pragma unroll
    for (int kb = 0; kb < 16; ++kb) {
      const int kk = kb * 32;
      v16bf Bf = cat16(*(const bf16x8*)(bbase + kk),  *(const bf16x8*)(bbase + kk + 8));
      v16bf A0 = cat16(*(const bf16x8*)(abase0 + kk), *(const bf16x8*)(abase0 + kk + 8));
      v16bf A1 = cat16(*(const bf16x8*)(abase1 + kk), *(const bf16x8*)(abase1 + kk + 8));
      acc0 = wmma_bf16(A0, Bf, acc0);
      acc1 = wmma_bf16(A1, Bf, acc1);
    }
    const float bias = sBeff[c0col + lrow];
    const int colg = c0col + lrow;
    const int rb = hi ? 8 : 0;
#pragma unroll
    for (int r = 0; r < 8; ++r) {
      sGates[(r + rb) * 128 + colg]        = acc0[r] + bias;
      sGates[(16 + r + rb) * 128 + colg]   = acc1[r] + bias;
    }
  };

  // out GEMM (waves 0..1): comb_h [32x512] @ W_lin slice [512x16] -> d_out[t]
  auto outGemm = [&](int tout, const bf16* sA) {
    const int m0 = wave * 16;
    const bf16* bbase = sWlin + lrow * 512 + (hi ? 16 : 0);
    const bf16* abase = sA + (m0 + lrow) * 512 + (hi ? 16 : 0);
    v8f acc = {};
#pragma unroll
    for (int kb = 0; kb < 16; ++kb) {
      const int kk = kb * 32;
      v16bf Bf = cat16(*(const bf16x8*)(bbase + kk), *(const bf16x8*)(bbase + kk + 8));
      v16bf Af = cat16(*(const bf16x8*)(abase + kk), *(const bf16x8*)(abase + kk + 8));
      acc = wmma_bf16(Af, Bf, acc);
    }
    const int oc = wg * 16 + lrow;
    const float bl = b_lin[oc];
    const int rb = hi ? 8 : 0;
#pragma unroll
    for (int r = 0; r < 8; ++r) {
      const int row = m0 + r + rb;
      out[((size_t)(grow0 + row) * T_ + tout) * D_ + oc] = acc[r] + bl;
    }
  };

  // ---- step 0 (gates precomputed) ----
  elementwise(0);
  group_barrier(myctr, 1);
  bf16* sAcur = sA0;
  bf16* sAnxt = sA1;
  asyncLoadA(sAcur, 0);
  wait_async0();
  __syncthreads();

  // ---- steps 1..T-1: one group barrier per step; async broadcast overlaps out-GEMM ----
  for (int t = 1; t < T_; ++t) {
    gatesGemm(sAcur);
    __syncthreads();
    elementwise(t);
    group_barrier(myctr, (unsigned)(t + 1));
    asyncLoadA(sAnxt, t & 1);          // DMA comb_h_t into the other tile ...
    if (wave < 2) outGemm(t - 1, sAcur); // ... while computing out_{t-1} from old tile
    wait_async0();
    __syncthreads();
    bf16* tmp = sAcur; sAcur = sAnxt; sAnxt = tmp;
  }
  if (wave < 2) outGemm(T_ - 1, sAcur);
}

// ---------------- host entry ----------------
extern "C" void kernel_launch(void* const* d_in, const int* in_sizes, int n_in,
                              void* d_out, int out_size, void* d_ws, size_t ws_size,
                              hipStream_t stream) {
  (void)in_sizes; (void)n_in; (void)out_size; (void)ws_size;
  const float* inp   = (const float*)d_in[0];
  const float* h0    = (const float*)d_in[1];
  const float* c0    = (const float*)d_in[2];
  const float* W_ih  = (const float*)d_in[3];
  const float* W_hh  = (const float*)d_in[4];
  const float* b_ih  = (const float*)d_in[5];
  const float* b_hh  = (const float*)d_in[6];
  const float* W_lin = (const float*)d_in[7];
  const float* b_lin = (const float*)d_in[8];
  const int*   w1    = (const int*)d_in[9];
  const int*   w2    = (const int*)d_in[10];
  float* out = (float*)d_out;

  char* ws = (char*)d_ws;
  unsigned* ctr   = (unsigned*)(ws + 0);                 //   2 KB (padded counters)
  bf16*   weffT   = (bf16*)  (ws + 4096);                //   2 MB
  float*  beff    = (float*) (ws + 4096 + 2097152);      //   8 KB
  bf16*   wlinT   = (bf16*)  (ws + 4096 + 2097152 + 8192);              // 256 KB
  float*  g0      = (float*) (ws + 4096 + 2097152 + 8192 + 262144);     //   2 MB
  bf16*   hc      = (bf16*)  (ws + 4096 + 2097152 + 8192 + 262144 + 2097152); // 512 KB

  k_zero  <<<dim3(1),    dim3(THREADS), 0, stream>>>(ctr);
  k_weff  <<<dim3(NG_),  dim3(THREADS), 0, stream>>>(W_ih, W_hh, W_lin, b_ih, b_hh, b_lin, weffT, beff);
  k_wlin  <<<dim3(512),  dim3(THREADS), 0, stream>>>(W_lin, wlinT);
  k_gates0<<<dim3(B_),   dim3(THREADS), 0, stream>>>(inp, h0, W_ih, W_hh, b_ih, b_hh, g0);

  constexpr size_t SMEM = (128*512 + 16*512 + 2*32*512) * sizeof(bf16)
                        + (32*128 + 2*8*32*32 + 32*32 + 128) * sizeof(float); // 299520
  rscn_main<<<dim3(NGROUPS * WPG), dim3(THREADS), SMEM, stream>>>(
      h0, c0, w1, w2, b_lin, weffT, beff, wlinT, g0, hc, ctr, out);
}
